// NextTokenGenerator_59940563583598
// MI455X (gfx1250) — matrix operand order repaired and checked by reading, not compiled, not run on hardware
//
#include <hip/hip_runtime.h>
#include <math.h>

// Problem constants (match reference): B=64, S=128, H=1024, V=50257, T=32 new tokens
#define BATCH   64
#define SEQ     128
#define HID     1024
#define VOCAB   50257
#define VPAD    50272          // VOCAB padded up to multiple of 16
#define GATES   4096           // 4*H
#define K2      2048           // concat(x,h) inner dim
#define NTOK    32

typedef __attribute__((ext_vector_type(16))) __bf16 v16bf;
typedef __attribute__((ext_vector_type(8)))  float  v8f;

// ---------- helpers ----------
__device__ __forceinline__ unsigned short f2bf(float f) {
    unsigned int u = __float_as_uint(f);
    u += 0x7FFFu + ((u >> 16) & 1u);   // round-to-nearest-even
    return (unsigned short)(u >> 16);
}
__device__ __forceinline__ float sigmoidf_(float x) {
    return 1.0f / (1.0f + __expf(-x));
}

// A fragment (16x32 bf16, MxK): lane<16 holds row M=lane, K = kb+[0..7] and kb+[16..23]
//                               lane>=16 holds row M=lane-16, K = kb+[8..15] and kb+[24..31]
__device__ __forceinline__ v16bf load_frag_a(const unsigned short* base, int stride,
                                             int row0, int kb, int lane) {
    int r = lane & 15, half = lane >> 4;
    const unsigned short* p = base + (size_t)(row0 + r) * stride + kb + half * 8;
    union { uint4 q[2]; v16bf v; } u;
    u.q[0] = *(const uint4*)(p);
    u.q[1] = *(const uint4*)(p + 16);
    return u.v;
}
// B fragment (32x16 bf16, KxN) built from row-major W (N rows, K contiguous):
// lane<16 holds col N=lane with K = kb+[0..7] and kb+[16..23]; lane>=16 holds col N=lane-16, K = kb+[8..15] and kb+[24..31]
__device__ __forceinline__ v16bf load_frag_b(const unsigned short* base, int stride,
                                             int nrow0, int kb, int lane) {
    int r = lane & 15, half = lane >> 4;
    const unsigned short* p = base + (size_t)(nrow0 + r) * stride + kb + half * 8;
    union { uint4 q[2]; v16bf v; } u;
    u.q[0] = *(const uint4*)(p);
    u.q[1] = *(const uint4*)(p + 16);
    return u.v;
}

#define WMMA_BF16(A_, B_, C_) \
    __builtin_amdgcn_wmma_f32_16x16x32_bf16(false, (A_), false, (B_), (short)0, (C_), false, false)

// One k-step worth of fragments: shared weight fragment + 4 M-tiles of A.
struct Frag5 { v16bf fb, fa0, fa1, fa2, fa3; };

__device__ __forceinline__ Frag5 load_set(const unsigned short* W, int wstride, int n16,
                                          const unsigned short* A, int astride,
                                          int kb, int lane) {
    Frag5 f;
    f.fb  = load_frag_b(W, wstride, n16, kb, lane);
    f.fa0 = load_frag_a(A, astride,  0, kb, lane);
    f.fa1 = load_frag_a(A, astride, 16, kb, lane);
    f.fa2 = load_frag_a(A, astride, 32, kb, lane);
    f.fa3 = load_frag_a(A, astride, 48, kb, lane);
    return f;
}
__device__ __forceinline__ void wmma4(const Frag5& f, v8f& a0, v8f& a1, v8f& a2, v8f& a3) {
    a0 = WMMA_BF16(f.fa0, f.fb, a0);
    a1 = WMMA_BF16(f.fa1, f.fb, a1);
    a2 = WMMA_BF16(f.fa2, f.fb, a2);
    a3 = WMMA_BF16(f.fa3, f.fb, a3);
}

// ---------- one-time packing ----------
__global__ void __launch_bounds__(256)
pack_wcat_kernel(const float* __restrict__ W_ih, const float* __restrict__ W_hh,
                 unsigned short* __restrict__ Wc) {
    int idx = blockIdx.x * 256 + threadIdx.x;     // GATES * K2 = 8M elements
    int n = idx >> 11, k = idx & (K2 - 1);
    float v = (k < HID) ? W_ih[(size_t)n * HID + k] : W_hh[(size_t)n * HID + (k - HID)];
    Wc[idx] = f2bf(v);
}

__global__ void __launch_bounds__(256)
pack_wfc_kernel(const float* __restrict__ W_fc, unsigned short* __restrict__ Wf) {
    size_t idx = (size_t)blockIdx.x * 256 + threadIdx.x;   // VPAD*HID
    if (idx >= (size_t)VPAD * HID) return;
    size_t n = idx >> 10;
    int k = (int)(idx & (HID - 1));
    Wf[idx] = (n < VOCAB) ? f2bf(W_fc[n * HID + k]) : (unsigned short)0;
}

__global__ void __launch_bounds__(64)
lengths_kernel(const int* __restrict__ ids, int* __restrict__ len) {
    int b = threadIdx.x;
    if (b < BATCH) {
        int n = 0;
        for (int s = 0; s < SEQ; ++s) n += (ids[b * SEQ + s] != 0);
        len[b] = n;
    }
}

__global__ void __launch_bounds__(256)
init_state_kernel(float* __restrict__ h, float* __restrict__ c,
                  unsigned short* __restrict__ A) {
    int idx = blockIdx.x * 256 + threadIdx.x;   // BATCH*HID
    h[idx] = 0.0f;
    c[idx] = 0.0f;
    int b = idx >> 10, j = idx & (HID - 1);
    A[(size_t)b * K2 + HID + j] = 0;            // bf16(h)=0
}

// ---------- per-step input assembly ----------
__global__ void __launch_bounds__(256)
build_x_prompt_kernel(const int* __restrict__ ids, const float* __restrict__ emb,
                      unsigned short* __restrict__ A, int t) {
    int idx = blockIdx.x * 256 + threadIdx.x;   // BATCH*HID
    int b = idx >> 10, j = idx & (HID - 1);
    int tok = ids[b * SEQ + t];
    A[(size_t)b * K2 + j] = f2bf(emb[(size_t)tok * HID + j]);
}

__global__ void __launch_bounds__(256)
build_x_decode_kernel(const int* __restrict__ tok, const float* __restrict__ emb,
                      unsigned short* __restrict__ A) {
    int idx = blockIdx.x * 256 + threadIdx.x;
    int b = idx >> 10, j = idx & (HID - 1);
    A[(size_t)b * K2 + j] = f2bf(emb[(size_t)tok[b] * HID + j]);
}

// ---------- gate GEMM: gates[64x4096] = A[64x2048] @ Wcat[4096x2048]^T + bias ----------
// One wave per N-tile; all 4 M-tiles accumulated in-wave (weight fragment fetched
// once). Ping-pong double-buffered k-loop: fragment sets alternate roles so there
// are no loop-carried register copies, and each WMMA group only waits on loads
// issued a half-iteration earlier.
__global__ void __launch_bounds__(256)
gate_gemm_kernel(const unsigned short* __restrict__ A, const unsigned short* __restrict__ W,
                 const float* __restrict__ b_ih, const float* __restrict__ b_hh,
                 float* __restrict__ gates) {
    const int wave  = threadIdx.x >> 5;
    const int lane  = threadIdx.x & 31;
    const int ntile = blockIdx.x * 8 + wave;    // 256 ntiles
    const int n16   = ntile * 16;

    v8f acc0 = {}, acc1 = {}, acc2 = {}, acc3 = {};

    Frag5 s0 = load_set(W, K2, n16, A, K2, 0, lane);
    Frag5 s1;
    int kb = 32;
#pragma unroll 1
    for (; kb < K2 - 32; kb += 64) {
        s1 = load_set(W, K2, n16, A, K2, kb, lane);
        wmma4(s0, acc0, acc1, acc2, acc3);
        s0 = load_set(W, K2, n16, A, K2, kb + 32, lane);
        wmma4(s1, acc0, acc1, acc2, acc3);
    }
    s1 = load_set(W, K2, n16, A, K2, kb, lane);     // kb == K2-32
    wmma4(s0, acc0, acc1, acc2, acc3);
    wmma4(s1, acc0, acc1, acc2, acc3);

    const int col  = n16 + (lane & 15);
    const int half = lane >> 4;
    const float bias = b_ih[col] + b_hh[col];
    v8f accs[4] = {acc0, acc1, acc2, acc3};
#pragma unroll
    for (int m = 0; m < 4; ++m) {
        float* gp = gates + (size_t)(m * 16 + half * 8) * GATES + col;
#pragma unroll
        for (int j = 0; j < 8; ++j) {
            gp[(size_t)j * GATES] = accs[m][j] + bias;
        }
    }
}

// ---------- LSTM cell pointwise update ----------
__global__ void __launch_bounds__(256)
lstm_cell_kernel(const float* __restrict__ gates, float* __restrict__ h, float* __restrict__ c,
                 unsigned short* __restrict__ A, const int* __restrict__ len, int t) {
    int idx = blockIdx.x * 256 + threadIdx.x;   // BATCH*HID
    int b = idx >> 10, j = idx & (HID - 1);
    bool valid = (t < 0) || (t < len[b]);

    const float* g = gates + (size_t)b * GATES;
    float iv = sigmoidf_(g[j]);
    float fv = sigmoidf_(g[HID + j]);
    float gv = tanhf(g[2 * HID + j]);
    float ov = sigmoidf_(g[3 * HID + j]);
    float cn = fv * c[idx] + iv * gv;
    float hn = ov * tanhf(cn);

    float hw;
    if (valid) { c[idx] = cn; h[idx] = hn; hw = hn; }
    else       { hw = h[idx]; }
    A[(size_t)b * K2 + HID + j] = f2bf(hw);     // bf16(h) for next gate GEMM + logits GEMM
}

// ---------- logits GEMM: out[:,s,:] = h[64x1024] @ Wfc[VPADx1024]^T + b_fc ----------
// Same ping-pong one-wave-per-N-tile structure: W_fc (103 MB bf16, L2-resident)
// is fetched once per 16-column tile and streamed behind the WMMAs.
__global__ void __launch_bounds__(256)
logits_gemm_kernel(const unsigned short* __restrict__ Ah /* A+HID, stride K2 */,
                   const unsigned short* __restrict__ Wf,
                   const float* __restrict__ b_fc, float* __restrict__ out, int s) {
    const int wave  = threadIdx.x >> 5;
    const int lane  = threadIdx.x & 31;
    const int NT    = VPAD / 16;                // 3142
    const int ntile = blockIdx.x * 8 + wave;
    if (ntile >= NT) return;
    const int n16 = ntile * 16;

    v8f acc0 = {}, acc1 = {}, acc2 = {}, acc3 = {};

    Frag5 s0 = load_set(Wf, HID, n16, Ah, K2, 0, lane);
    Frag5 s1;
    int kb = 32;
#pragma unroll 1
    for (; kb < HID - 32; kb += 64) {
        s1 = load_set(Wf, HID, n16, Ah, K2, kb, lane);
        wmma4(s0, acc0, acc1, acc2, acc3);
        s0 = load_set(Wf, HID, n16, Ah, K2, kb + 32, lane);
        wmma4(s1, acc0, acc1, acc2, acc3);
    }
    s1 = load_set(Wf, HID, n16, Ah, K2, kb, lane);  // kb == HID-32
    wmma4(s0, acc0, acc1, acc2, acc3);
    wmma4(s1, acc0, acc1, acc2, acc3);

    const int col = n16 + (lane & 15);
    if (col >= VOCAB) return;
    const int half = lane >> 4;
    const float bias = b_fc[col];
    v8f accs[4] = {acc0, acc1, acc2, acc3};
#pragma unroll
    for (int m = 0; m < 4; ++m) {
#pragma unroll
        for (int j = 0; j < 8; ++j) {
            int row = m * 16 + half * 8 + j;    // batch index
            out[((size_t)row * NTOK + s) * VOCAB + col] = accs[m][j] + bias;
        }
    }
}

// ---------- argmax over previous logits slice ----------
__global__ void __launch_bounds__(256)
argmax_kernel(const float* __restrict__ out, int s_prev, int* __restrict__ tok) {
    int b = blockIdx.x;
    const float* row = out + ((size_t)b * NTOK + s_prev) * VOCAB;
    __shared__ float smax[256];
    __shared__ int   sidx[256];
    float best = -INFINITY; int bi = 0;
    for (int v = threadIdx.x; v < VOCAB; v += 256) {
        float x = row[v];
        if (x > best) { best = x; bi = v; }
    }
    smax[threadIdx.x] = best; sidx[threadIdx.x] = bi;
    __syncthreads();
    for (int off = 128; off > 0; off >>= 1) {
        if (threadIdx.x < off) {
            float o = smax[threadIdx.x + off]; int oi = sidx[threadIdx.x + off];
            if (o > smax[threadIdx.x] ||
                (o == smax[threadIdx.x] && oi < sidx[threadIdx.x])) {
                smax[threadIdx.x] = o; sidx[threadIdx.x] = oi;
            }
        }
        __syncthreads();
    }
    if (threadIdx.x == 0) tok[b] = sidx[0];
}

// ---------- launcher ----------
extern "C" void kernel_launch(void* const* d_in, const int* in_sizes, int n_in,
                              void* d_out, int out_size, void* d_ws, size_t ws_size,
                              hipStream_t stream) {
    const int*   input_ids = (const int*)  d_in[0];
    /* d_in[1] = max_new_tokens (fixed = 32) */
    const float* emb  = (const float*)d_in[2];
    const float* W_ih = (const float*)d_in[3];
    const float* W_hh = (const float*)d_in[4];
    const float* b_ih = (const float*)d_in[5];
    const float* b_hh = (const float*)d_in[6];
    const float* W_fc = (const float*)d_in[7];
    const float* b_fc = (const float*)d_in[8];
    float* out = (float*)d_out;

    // workspace carve-out (256B aligned)
    char* p = (char*)d_ws;
    auto carve = [&](size_t bytes) -> void* {
        void* r = (void*)p;
        p += (bytes + 255) & ~(size_t)255;
        return r;
    };
    unsigned short* Wc  = (unsigned short*)carve((size_t)GATES * K2 * 2);   //  16.8 MB bf16 gate weights
    unsigned short* Wf  = (unsigned short*)carve((size_t)VPAD * HID * 2);   // 103   MB bf16 fc weights (L2-resident)
    unsigned short* A   = (unsigned short*)carve((size_t)BATCH * K2 * 2);   // bf16 concat(x, h)
    float*          gts = (float*)carve((size_t)BATCH * GATES * 4);
    float*          h   = (float*)carve((size_t)BATCH * HID * 4);
    float*          c   = (float*)carve((size_t)BATCH * HID * 4);
    int*            len = (int*)carve(BATCH * 4);
    int*            tok = (int*)carve(BATCH * 4);
    (void)ws_size; (void)n_in; (void)in_sizes; (void)out_size;

    const int BH_BLOCKS   = (BATCH * HID) / 256;                 // 256
    const int GATE_BLOCKS = (GATES / 16) / 8;                    // 32  (one wave per ntile)
    const int LOG_BLOCKS  = ((VPAD / 16) + 7) / 8;               // 393 (one wave per ntile)

    // one-time weight packing + state init
    pack_wcat_kernel<<<(GATES * K2) / 256, 256, 0, stream>>>(W_ih, W_hh, Wc);
    pack_wfc_kernel<<<(int)(((size_t)VPAD * HID + 255) / 256), 256, 0, stream>>>(W_fc, Wf);
    lengths_kernel<<<1, 64, 0, stream>>>(input_ids, len);
    init_state_kernel<<<BH_BLOCKS, 256, 0, stream>>>(h, c, A);

    // prompt phase: 128 masked LSTM steps
    for (int t = 0; t < SEQ; ++t) {
        build_x_prompt_kernel<<<BH_BLOCKS, 256, 0, stream>>>(input_ids, emb, A, t);
        gate_gemm_kernel<<<GATE_BLOCKS, 256, 0, stream>>>(A, Wc, b_ih, b_hh, gts);
        lstm_cell_kernel<<<BH_BLOCKS, 256, 0, stream>>>(gts, h, c, A, len, t);
    }

    // logits for step 0
    logits_gemm_kernel<<<LOG_BLOCKS, 256, 0, stream>>>(A + HID, Wf, b_fc, out, 0);

    // decode phase: 31 more steps (argmax of previous logits -> embed -> LSTM -> logits)
    for (int s = 1; s < NTOK; ++s) {
        argmax_kernel<<<BATCH, 256, 0, stream>>>(out, s - 1, tok);
        build_x_decode_kernel<<<BH_BLOCKS, 256, 0, stream>>>(tok, emb, A);
        gate_gemm_kernel<<<GATE_BLOCKS, 256, 0, stream>>>(A, Wc, b_ih, b_hh, gts);
        lstm_cell_kernel<<<BH_BLOCKS, 256, 0, stream>>>(gts, h, c, A, len, -1);
        logits_gemm_kernel<<<LOG_BLOCKS, 256, 0, stream>>>(A + HID, Wf, b_fc, out, s);
    }
}
